// JambaSparseMoeBlock_75771813036250
// MI455X (gfx1250) — matrix operand order repaired and checked by reading, not compiled, not run on hardware
//
#include <hip/hip_runtime.h>

typedef __attribute__((ext_vector_type(16))) __bf16 v16bf;
typedef __attribute__((ext_vector_type(8)))  __bf16 v8bf;
typedef __attribute__((ext_vector_type(4)))  __bf16 v4bf;
typedef __attribute__((ext_vector_type(8)))  float  v8f;

#define T_TOK 32768
#define D_DIM 1024
#define F_DIM 2048
#define E_NUM 16

__device__ inline v8f wmma_bf16(v16bf a, v16bf b, v8f c) {
  return __builtin_amdgcn_wmma_f32_16x16x32_bf16(false, a, false, b, (short)0, c,
                                                 false, false);
}

// A fragment: lane holds row M=lane&15; K chunk0 = k0+koff..+7, chunk1 = k0+16+koff..+7
__device__ inline v16bf ld_a_contig(const __bf16* p) {
  v8bf lo = *(const v8bf*)p;
  v8bf hi = *(const v8bf*)(p + 16);
  v16bf r;
#pragma unroll
  for (int i = 0; i < 8; ++i) { r[i] = lo[i]; r[i + 8] = hi[i]; }
  return r;
}

__device__ inline v16bf cat8(v8bf lo, v8bf hi) {
  v16bf r;
#pragma unroll
  for (int i = 0; i < 8; ++i) { r[i] = lo[i]; r[i + 8] = hi[i]; }
  return r;
}

// ---------------- fp32 -> bf16 row-major copy (x) ----------------
__global__ __launch_bounds__(256) void cvt_x_kernel(const float* __restrict__ in,
                                                    __bf16* __restrict__ out) {
  size_t i = ((size_t)blockIdx.x * 256 + threadIdx.x) * 4;
  float4 v = *(const float4*)(in + i);
  v4bf o = { (__bf16)v.x, (__bf16)v.y, (__bf16)v.z, (__bf16)v.w };
  *(v4bf*)(out + i) = o;
}

// ------------- fp32 [R][C] -> bf16 [C][R] per expert (blockIdx.z) -------------
__global__ __launch_bounds__(256) void transpose_cvt_kernel(const float* __restrict__ in,
                                                            __bf16* __restrict__ out,
                                                            int R, int C) {
  __shared__ float tile[32][33];
  const float* ip = in + (size_t)blockIdx.z * R * C;
  __bf16* op = out + (size_t)blockIdx.z * R * C;
  int c0 = blockIdx.x * 32, r0 = blockIdx.y * 32;
  int tx = threadIdx.x, ty = threadIdx.y;  // 32 x 8
#pragma unroll
  for (int i = 0; i < 32; i += 8)
    tile[ty + i][tx] = ip[(size_t)(r0 + ty + i) * C + (c0 + tx)];
  __syncthreads();
#pragma unroll
  for (int i = 0; i < 32; i += 8)
    op[(size_t)(c0 + ty + i) * R + (r0 + tx)] = (__bf16)tile[tx][ty + i];
}

// ---------------- router: logits, softmax, top-2, expert lists ----------------
__global__ __launch_bounds__(256) void router_kernel(
    const float* __restrict__ x, const float* __restrict__ rw,
    float* __restrict__ logits_out, int* __restrict__ cnt,
    int* __restrict__ tokList, float* __restrict__ wList) {
  int lane = threadIdx.x & 31, wv = threadIdx.x >> 5;
  int t = blockIdx.x * 8 + wv;
  const float* xp = x + (size_t)t * D_DIM;
  float acc[E_NUM];
#pragma unroll
  for (int e = 0; e < E_NUM; ++e) acc[e] = 0.0f;
  for (int d = lane; d < D_DIM; d += 32) {
    float xv = xp[d];
#pragma unroll
    for (int e = 0; e < E_NUM; ++e) acc[e] += xv * rw[e * D_DIM + d];
  }
#pragma unroll
  for (int e = 0; e < E_NUM; ++e)
#pragma unroll
    for (int off = 16; off > 0; off >>= 1) acc[e] += __shfl_xor(acc[e], off, 32);

  if (lane == 0) {
    float mx = acc[0];
#pragma unroll
    for (int e = 1; e < E_NUM; ++e) mx = fmaxf(mx, acc[e]);
    float p[E_NUM], s = 0.0f;
#pragma unroll
    for (int e = 0; e < E_NUM; ++e) { p[e] = __expf(acc[e] - mx); s += p[e]; }
    int i1 = 0;
#pragma unroll
    for (int e = 1; e < E_NUM; ++e) if (p[e] > p[i1]) i1 = e;
    int i2 = (i1 == 0) ? 1 : 0;
#pragma unroll
    for (int e = 0; e < E_NUM; ++e) if (e != i1 && p[e] > p[i2]) i2 = e;
    float inv = 1.0f / s;
    float w1 = p[i1] * inv, w2 = p[i2] * inv;  // softmax probs, no re-norm (Jamba)
    float* lo = logits_out + (size_t)t * E_NUM;
#pragma unroll
    for (int e = 0; e < E_NUM; ++e) lo[e] = acc[e];
    int p1 = atomicAdd(&cnt[i1], 1);
    tokList[i1 * T_TOK + p1] = t; wList[i1 * T_TOK + p1] = w1;
    int p2 = atomicAdd(&cnt[i2], 1);
    tokList[i2 * T_TOK + p2] = t; wList[i2 * T_TOK + p2] = w2;
  }
}

// ---- fused expert GEMM: silu(X G) * (X U) -> H (LDS); H @ down -> out (atomic) ----
__global__ __launch_bounds__(256) void moe_expert_kernel(
    const __bf16* __restrict__ xbf, const __bf16* __restrict__ gate_t,
    const __bf16* __restrict__ up_t, const __bf16* __restrict__ down_t,
    const int* __restrict__ cnt, const int* __restrict__ tokList,
    const float* __restrict__ wList, float* __restrict__ out) {
  const int e = blockIdx.y;
  const int m0 = blockIdx.x * 16;
  const int n = cnt[e];
  if (m0 >= n) return;  // block-uniform: EXEC stays all-ones for WMMA

  __shared__ __bf16 Hs[16 * F_DIM];  // 64KB, XOR-swizzled in 8-elem groups

  const int lane = threadIdx.x & 31, wv = threadIdx.x >> 5;
  const int hf = lane >> 4, mn = lane & 15;
  const int koff = hf * 8;

  const __bf16* ga = gate_t + (size_t)e * F_DIM * D_DIM;
  const __bf16* ua = up_t   + (size_t)e * F_DIM * D_DIM;
  const __bf16* da = down_t + (size_t)e * D_DIM * F_DIM;
  const int* tl = tokList + (size_t)e * T_TOK;
  const float* wl = wList + (size_t)e * T_TOK;

  const int mIdx = m0 + mn;
  const int tokA = (mIdx < n) ? tl[mIdx] : tl[0];
  const __bf16* arow = xbf + (size_t)tokA * D_DIM;

  // -------- Phase 1: H[16][F], two N-tiles per iteration --------
  for (int tt = 0; tt < 8; ++tt) {
    const int fn0 = (wv * 16 + tt * 2) * 16;             // tile 0; tile 1 = fn0+16
    const __bf16* bg0 = ga + (size_t)(fn0 + mn) * D_DIM + hf * 16;
    const __bf16* bu0 = ua + (size_t)(fn0 + mn) * D_DIM + hf * 16;
    const __bf16* bg1 = bg0 + (size_t)16 * D_DIM;
    const __bf16* bu1 = bu0 + (size_t)16 * D_DIM;
    v8f cg0 = {}, cu0 = {}, cg1 = {}, cu1 = {};
    for (int k0 = 0; k0 < D_DIM; k0 += 32) {
      v16bf a   = *(const v16bf*)(arow + k0 + hf * 16);
      v16bf vg0 = *(const v16bf*)(bg0 + k0);  // 16 contiguous K (transposed weights)
      v16bf vu0 = *(const v16bf*)(bu0 + k0);
      v16bf vg1 = *(const v16bf*)(bg1 + k0);
      v16bf vu1 = *(const v16bf*)(bu1 + k0);
      cg0 = wmma_bf16(a, vg0, cg0);
      cu0 = wmma_bf16(a, vu0, cu0);
      cg1 = wmma_bf16(a, vg1, cg1);
      cu1 = wmma_bf16(a, vu1, cu1);
    }
#pragma unroll
    for (int r = 0; r < 8; ++r) {
      const int row = r + hf * 8;               // C/D layout: VGPR r -> M=r / M=8+r
      float g0 = cg0[r];
      float h0 = g0 * __frcp_rn(1.0f + __expf(-g0)) * cu0[r];
      float g1 = cg1[r];
      float h1 = g1 * __frcp_rn(1.0f + __expf(-g1)) * cu1[r];
      const int f0 = fn0 + mn, f1 = fn0 + 16 + mn;
      Hs[row * F_DIM + ((((f0 >> 3) ^ row) << 3) | (f0 & 7))] = (__bf16)h0;
      Hs[row * F_DIM + ((((f1 >> 3) ^ row) << 3) | (f1 & 7))] = (__bf16)h1;
    }
  }
  __syncthreads();

  // -------- Phase 2: out += w * (H @ down), two N-tiles per iteration --------
  const int rowA = mn;
  for (int tt = 0; tt < 4; ++tt) {
    const int dn0 = (wv * 8 + tt * 2) * 16;              // tile 0; tile 1 = dn0+16
    const __bf16* bd0 = da + (size_t)(dn0 + mn) * F_DIM + hf * 16;
    const __bf16* bd1 = bd0 + (size_t)16 * F_DIM;
    v8f c0 = {}, c1 = {};
    for (int k0 = 0; k0 < F_DIM; k0 += 32) {
      const int klo = k0 + hf * 16, khi = klo + 8;
      v8bf alo = *(const v8bf*)&Hs[rowA * F_DIM + (((klo >> 3) ^ rowA) << 3)];
      v8bf ahi = *(const v8bf*)&Hs[rowA * F_DIM + (((khi >> 3) ^ rowA) << 3)];
      v16bf a = cat8(alo, ahi);
      v16bf b0 = *(const v16bf*)(bd0 + k0);
      v16bf b1 = *(const v16bf*)(bd1 + k0);
      c0 = wmma_bf16(a, b0, c0);
      c1 = wmma_bf16(a, b1, c1);
    }
#pragma unroll
    for (int r = 0; r < 8; ++r) {
      const int row = r + hf * 8;
      const int m2 = m0 + row;
      float w = (m2 < n) ? wl[m2] : 0.0f;
      int trow = (m2 < n) ? tl[m2] : 0;
      float* orow = out + (size_t)trow * D_DIM + mn;
      atomicAdd(orow + dn0, c0[r] * w);
      atomicAdd(orow + dn0 + 16, c1[r] * w);
    }
  }
}

extern "C" void kernel_launch(void* const* d_in, const int* in_sizes, int n_in,
                              void* d_out, int out_size, void* d_ws, size_t ws_size,
                              hipStream_t stream) {
  const float* x        = (const float*)d_in[0];  // [T, D]
  const float* router_w = (const float*)d_in[1];  // [E, D]
  const float* gate_w   = (const float*)d_in[2];  // [E, D, F]
  const float* up_w     = (const float*)d_in[3];  // [E, D, F]
  const float* down_w   = (const float*)d_in[4];  // [E, F, D]
  float* outp = (float*)d_out;                    // [T*D] out ++ [T*E] logits
  float* logits = outp + (size_t)T_TOK * D_DIM;

  // workspace carve-up
  char* ws = (char*)d_ws;
  const size_t szTD = (size_t)T_TOK * D_DIM * sizeof(__bf16);      // 64 MiB
  const size_t szW  = (size_t)E_NUM * D_DIM * F_DIM * sizeof(__bf16);
  __bf16* xbf    = (__bf16*)(ws);                 ws += szTD;
  __bf16* gate_t = (__bf16*)(ws);                 ws += szW;
  __bf16* up_t   = (__bf16*)(ws);                 ws += szW;
  __bf16* down_t = (__bf16*)(ws);                 ws += szW;
  int*    cntp   = (int*)(ws);                    ws += 256;
  int*    tokL   = (int*)(ws);                    ws += (size_t)E_NUM * T_TOK * 4;
  float*  wL     = (float*)(ws);

  hipMemsetAsync(d_out, 0, (size_t)T_TOK * D_DIM * sizeof(float), stream);
  hipMemsetAsync(cntp, 0, 256, stream);

  cvt_x_kernel<<<(T_TOK * (size_t)D_DIM) / (256 * 4), 256, 0, stream>>>(x, xbf);

  dim3 tb(32, 8);
  transpose_cvt_kernel<<<dim3(F_DIM / 32, D_DIM / 32, E_NUM), tb, 0, stream>>>(
      gate_w, gate_t, D_DIM, F_DIM);
  transpose_cvt_kernel<<<dim3(F_DIM / 32, D_DIM / 32, E_NUM), tb, 0, stream>>>(
      up_w, up_t, D_DIM, F_DIM);
  transpose_cvt_kernel<<<dim3(D_DIM / 32, F_DIM / 32, E_NUM), tb, 0, stream>>>(
      down_w, down_t, F_DIM, D_DIM);

  router_kernel<<<T_TOK / 8, 256, 0, stream>>>(x, router_w, logits, cntp, tokL, wL);

  moe_expert_kernel<<<dim3(T_TOK / 16, E_NUM), 256, 0, stream>>>(
      xbf, gate_t, up_t, down_t, cntp, tokL, wL, outp);
}
